// CausalSelfAttention_72241349918920
// MI455X (gfx1250) — matrix operand order repaired and checked by reading, not compiled, not run on hardware
//
#include <hip/hip_runtime.h>
#include <hip/hip_bf16.h>
#include <math.h>

// ---------------------------------------------------------------------------
// CausalSelfAttention (banded, window=64) for MI455X / gfx1250 (wave32, WMMA)
//   B=2, T=2048, C=1024, H=16, D=64.
//   Pipeline:
//     0a) xh   = f16(x)                       (elementwise)
//     0b) Wt   = f16(W^T) for Wqkv, Wproj     (LDS-tiled transpose+convert)
//     1) qkvh = f16(xh @ Wqkv + bqkv)         (WMMA GEMM, async LDS staging)
//     2) banded attention -> yh (f16)         (WMMA, S^T softmax-in-regs)
//     3) out  = yh @ Wproj + bproj (f32)      (same WMMA GEMM)
// ---------------------------------------------------------------------------

typedef _Float16 h4v  __attribute__((ext_vector_type(4)));
typedef _Float16 h8v  __attribute__((ext_vector_type(8)));
typedef _Float16 h16v __attribute__((ext_vector_type(16)));
typedef float    f4v  __attribute__((ext_vector_type(4)));
typedef float    v8f  __attribute__((ext_vector_type(8)));

static __device__ __forceinline__ v8f vzero8() {
    v8f z;
#pragma unroll
    for (int i = 0; i < 8; ++i) z[i] = 0.0f;
    return z;
}

static __device__ __forceinline__ h16v cat8(h8v lo, h8v hi) {
    return __builtin_shufflevector(lo, hi, 0, 1, 2, 3, 4, 5, 6, 7,
                                   8, 9, 10, 11, 12, 13, 14, 15);
}

static __device__ __forceinline__ v8f wmma_f16(h16v a, h16v b, v8f c) {
    // D = A(16x32 f16) * B(32x16 f16) + C(16x16 f32)
    return __builtin_amdgcn_wmma_f32_16x16x32_f16(
        /*neg_a=*/false, a, /*neg_b=*/false, b,
        /*c_mod=*/(short)0, c, /*reuse_a=*/false, /*reuse_b=*/false);
}

#define LD8(p) (*(const h8v*)(p))

// CDNA5 async global->LDS copy, 16 bytes per lane. ASYNCcnt-tracked.
// LDS operand = low 32 bits of the generic pointer (ISA 10.2: flat LDS
// addresses truncate to the wave-relative LDS byte address).
static __device__ __forceinline__ void async_cp16(const _Float16* g, _Float16* l) {
    const unsigned laddr = (unsigned)(unsigned long long)l;
    asm volatile("global_load_async_to_lds_b128 %0, %1, off"
                 :: "v"(laddr), "v"(g) : "memory");
}
static __device__ __forceinline__ void async_wait0() {
    asm volatile("s_wait_asynccnt 0x0" ::: "memory");
}

// ---------------------------------------------------------------------------
// 0a) f32 -> f16 elementwise (n4 = #float4 quads).
// ---------------------------------------------------------------------------
__global__ __launch_bounds__(256) void cvt_f16_kernel(
    const float* __restrict__ in, _Float16* __restrict__ out, int n4)
{
    const int i = blockIdx.x * 256 + threadIdx.x;
    if (i < n4) {
        f4v v = *(const f4v*)&in[(size_t)i * 4];
        *(h4v*)&out[(size_t)i * 4] = __builtin_convertvector(v, h4v);
    }
}

// ---------------------------------------------------------------------------
// 0b) Wt[N][K] (f16) = transpose(W[K][N] f32). 64x64 tiles, 256 threads.
// ---------------------------------------------------------------------------
__global__ __launch_bounds__(256) void transpose_cvt_kernel(
    const float* __restrict__ W, _Float16* __restrict__ Wt, int K, int N)
{
    __shared__ float t[64][65];
    const int tid = threadIdx.x;
    const int kb = blockIdx.y * 64, nb = blockIdx.x * 64;
#pragma unroll
    for (int i = 0; i < 4; ++i) {
        int q = tid + 256 * i;                 // 1024 float4 quads
        int kr = q >> 4, nq = (q & 15) << 2;
        f4v v = *(const f4v*)&W[(size_t)(kb + kr) * N + nb + nq];
        t[kr][nq + 0] = v[0]; t[kr][nq + 1] = v[1];
        t[kr][nq + 2] = v[2]; t[kr][nq + 3] = v[3];
    }
    __syncthreads();
#pragma unroll
    for (int i = 0; i < 4; ++i) {
        int q = tid + 256 * i;
        int nr = q >> 4, kq = (q & 15) << 2;
        h4v h = { (_Float16)t[kq + 0][nr], (_Float16)t[kq + 1][nr],
                  (_Float16)t[kq + 2][nr], (_Float16)t[kq + 3][nr] };
        *(h4v*)&Wt[(size_t)(nb + nr) * K + kb + kq] = h;
    }
}

// ---------------------------------------------------------------------------
// GEMM: C[M,N] = A[M,K](f16) * Wt[N][K](f16)^T + bias[N].  128x128x64 tiles,
// 256 threads (8 waves); wave w owns a 32x64 slice (2x4 WMMA accumulators).
// Tiles staged via global_load_async_to_lds_b128.
// ---------------------------------------------------------------------------
template <bool OUT_F16>
__global__ __launch_bounds__(256) void gemm_bias_wmma(
    const _Float16* __restrict__ A, const _Float16* __restrict__ Wt,
    const float* __restrict__ bias, void* __restrict__ Cout,
    int M, int N, int K)
{
    __shared__ _Float16 lA[128][72];   // [m][k]
    __shared__ _Float16 lB[128][72];   // [n][k]

    const int tid  = threadIdx.x;
    const int lane = tid & 31;
    const int w    = tid >> 5;
    const int ln   = lane & 15;
    const int lh   = lane >> 4;

    const int mBase = blockIdx.y * 128;
    const int nBase = blockIdx.x * 128;
    const int moff  = (w >> 1) * 32;   // wave rows (4 groups)
    const int noff  = (w & 1) * 64;    // wave cols (2 groups)

    v8f acc[2][4];
#pragma unroll
    for (int mi = 0; mi < 2; ++mi)
#pragma unroll
        for (int ni = 0; ni < 4; ++ni) acc[mi][ni] = vzero8();

    for (int k0 = 0; k0 < K; k0 += 64) {
        // Stage A and B tiles: 128 rows x 64 halfs = 8 chunks/row, 1024 each.
#pragma unroll
        for (int i = 0; i < 4; ++i) {
            int idx = tid + 256 * i;
            int row = idx >> 3, c = (idx & 7) << 3;
            async_cp16(&A [(size_t)(mBase + row) * K + k0 + c], &lA[row][c]);
            async_cp16(&Wt[(size_t)(nBase + row) * K + k0 + c], &lB[row][c]);
        }
        async_wait0();
        __syncthreads();

#pragma unroll
        for (int ks = 0; ks < 2; ++ks) {
            const int ao = ks * 32 + 8 * lh;   // A-frag: k interleaved 8-blocks
            h16v a0 = cat8(LD8(&lA[moff + ln][ao]),      LD8(&lA[moff + ln][ao + 16]));
            h16v a1 = cat8(LD8(&lA[moff + 16 + ln][ao]), LD8(&lA[moff + 16 + ln][ao + 16]));
            const int bo = ks * 32 + 8 * lh;   // B-frag: k interleaved 8-blocks
#pragma unroll
            for (int ni = 0; ni < 4; ++ni) {
                h16v b = cat8(LD8(&lB[noff + ni * 16 + ln][bo]),
                              LD8(&lB[noff + ni * 16 + ln][bo + 16]));
                acc[0][ni] = wmma_f16(a0, b, acc[0][ni]);
                acc[1][ni] = wmma_f16(a1, b, acc[1][ni]);
            }
        }
        __syncthreads();
    }

    // Epilogue: C-layout lane l holds (m = r + 8*lh, n = ln) per tile.
#pragma unroll
    for (int ni = 0; ni < 4; ++ni) {
        const int n = nBase + noff + ni * 16 + ln;
        const float bv = bias[n];
#pragma unroll
        for (int mi = 0; mi < 2; ++mi)
#pragma unroll
            for (int r = 0; r < 8; ++r) {
                const int m = mBase + moff + mi * 16 + r + 8 * lh;
                if constexpr (OUT_F16)
                    ((_Float16*)Cout)[(size_t)m * N + n] = (_Float16)(acc[mi][ni][r] + bv);
                else
                    ((float*)Cout)[(size_t)m * N + n] = acc[mi][ni][r] + bv;
            }
    }
}

// ---------------------------------------------------------------------------
// Banded attention. One block = (b, head, 64-query tile), 128 threads/4 waves.
// qkv is f16 [B,T,3C]. Keys span [qbase-64, qbase+63].  S^T = K*Q^T so each
// query's key band lives in one lane pair; softmax = regs + shfl_xor(16); the
// normalized P is already in A-fragment layout for Y = P*V.  y written f16.
// ---------------------------------------------------------------------------
#define AT_T 2048
#define AT_C 1024
#define AT_D 64

__global__ __launch_bounds__(128) void attn_band_wmma(
    const _Float16* __restrict__ qkv,  // [B,T,3C] f16
    _Float16* __restrict__ y)          // [B,T,C]  f16
{
    __shared__ _Float16 Qs[64][72];    // [q][d]
    __shared__ _Float16 Ks[128][72];   // [key][d]
    __shared__ _Float16 Vt[64][136];   // [d][key] (V transposed)

    const int tid  = threadIdx.x;
    const int lane = tid & 31;
    const int wq   = tid >> 5;    // wave id: 16 queries each
    const int ln   = lane & 15;
    const int lh   = lane >> 4;

    const int bid   = blockIdx.x;
    const int qt    = bid & 31;          // T/64 = 32 tiles
    const int hd    = (bid >> 5) & 15;   // head
    const int b     = bid >> 9;
    const int qbase = qt * 64;
    const int kbase = qbase - 64;

    const size_t bOff = (size_t)b * AT_T;
    const _Float16* qp = qkv + bOff * 3 * AT_C + hd * AT_D;  // row stride 3C
    const _Float16* kp = qp + AT_C;
    const _Float16* vp = qp + 2 * AT_C;

    // Stage Q (64 rows x 64 halfs, async): 512 16B chunks over 128 threads.
#pragma unroll
    for (int i = 0; i < 4; ++i) {
        int idx = tid + 128 * i;
        int row = idx >> 3, c = (idx & 7) << 3;
        async_cp16(&qp[(size_t)(qbase + row) * 3 * AT_C + c], &Qs[row][c]);
    }
    // Stage K (128 rows, async), clamp negative rows (masked later).
#pragma unroll
    for (int i = 0; i < 8; ++i) {
        int idx = tid + 128 * i;
        int row = idx >> 3, c = (idx & 7) << 3;
        int t = kbase + row; if (t < 0) t = 0;
        async_cp16(&kp[(size_t)t * 3 * AT_C + c], &Ks[row][c]);
    }
    // Stage V transposed (Vt[d][key]) - manual (transpose on store).
#pragma unroll
    for (int i = 0; i < 16; ++i) {
        int idx = tid + 128 * i;               // 128 rows x 16 d-quads
        int row = idx >> 4, dq = (idx & 15) << 2;
        int t = kbase + row; if (t < 0) t = 0;
        h4v v = *(const h4v*)&vp[(size_t)t * 3 * AT_C + dq];
        Vt[dq + 0][row] = v[0];
        Vt[dq + 1][row] = v[1];
        Vt[dq + 2][row] = v[2];
        Vt[dq + 3][row] = v[3];
    }
    async_wait0();
    __syncthreads();

    // ---- GEMM1: S^T[128 keys x 16 q] = K * Q^T (8 m-tiles, K-dim = D = 64).
    v8f s[8];
#pragma unroll
    for (int t = 0; t < 8; ++t) s[t] = vzero8();

#pragma unroll
    for (int ks = 0; ks < 2; ++ks) {
        const int bo = ks * 32 + 8 * lh;                   // B = Q^T frag
        h16v bq = cat8(LD8(&Qs[wq * 16 + ln][bo]), LD8(&Qs[wq * 16 + ln][bo + 16]));
        const int ao = ks * 32 + 8 * lh;                   // A = K frag
#pragma unroll
        for (int t = 0; t < 8; ++t) {
            h16v ak = cat8(LD8(&Ks[t * 16 + ln][ao]), LD8(&Ks[t * 16 + ln][ao + 16]));
            s[t] = wmma_f16(ak, bq, s[t]);
        }
    }

    // ---- Mask + scale + softmax.
    // Thread's element (t,r) is key = base + e,  e = 16t + r (constant when
    // unrolled), base = kbase + 8*lh.  Visible iff 0 <= key <= q, key >= q-64,
    // i.e. e in [lo, lo+range]: a single unsigned range test per element.
    const int q    = qbase + wq * 16 + ln;
    const int base = kbase + 8 * lh;
    int lo = q - 64 - base;
    if (lo < -base) lo = -base;                    // enforce key >= 0
    const unsigned range = (unsigned)(q - base - lo);
    const float NEG = -3.0e38f;
    float mx = NEG;
#pragma unroll
    for (int t = 0; t < 8; ++t)
#pragma unroll
        for (int r = 0; r < 8; ++r) {
            const int e = 16 * t + r;
            const bool vis = (unsigned)(e - lo) <= range;
            const float val = vis ? s[t][r] * 0.125f : NEG;   // 1/sqrt(64)
            s[t][r] = val;
            mx = fmaxf(mx, val);
        }
    mx = fmaxf(mx, __shfl_xor(mx, 16, 32));
    float sum = 0.0f;
#pragma unroll
    for (int t = 0; t < 8; ++t)
#pragma unroll
        for (int r = 0; r < 8; ++r) {
            const float e = __expf(s[t][r] - mx);
            s[t][r] = e;
            sum += e;
        }
    sum += __shfl_xor(sum, 16, 32);
    const float inv = 1.0f / sum;

    // ---- P A-fragments directly from registers.
    h16v pa[4];
#pragma unroll
    for (int st = 0; st < 4; ++st)
#pragma unroll
        for (int e = 0; e < 16; ++e)
            pa[st][e] = (_Float16)(s[2 * st + (e >> 3)][e & 7] * inv);

    // ---- GEMM2: Y[16 q x 64 d] = P * V (4 n-tiles, 4 k-steps of 32 keys).
#pragma unroll
    for (int dt = 0; dt < 4; ++dt) {
        v8f yac = vzero8();
#pragma unroll
        for (int st = 0; st < 4; ++st) {
            const int ko = st * 32 + 8 * lh;
            h16v bv = cat8(LD8(&Vt[dt * 16 + ln][ko]), LD8(&Vt[dt * 16 + ln][ko + 16]));
            yac = wmma_f16(pa[st], bv, yac);
        }
#pragma unroll
        for (int r = 0; r < 8; ++r) {
            const int qq = qbase + wq * 16 + r + 8 * lh;
            y[(bOff + qq) * AT_C + hd * AT_D + dt * 16 + ln] = (_Float16)yac[r];
        }
    }
}

// ---------------------------------------------------------------------------
// Launch.  Inputs: x[B,T,C], Wqkv[C,3C], bqkv[3C], Wproj[C,C], bproj[C].
// Workspace (f16): xh 8.4MB, Wqkvt 6.3MB, Wprojt 2.1MB, qkvh 25.2MB, yh 8.4MB.
// ---------------------------------------------------------------------------
extern "C" void kernel_launch(void* const* d_in, const int* in_sizes, int n_in,
                              void* d_out, int out_size, void* d_ws, size_t ws_size,
                              hipStream_t stream)
{
    (void)in_sizes; (void)n_in; (void)out_size; (void)ws_size;
    const float* x     = (const float*)d_in[0];
    const float* Wqkv  = (const float*)d_in[1];
    const float* bqkv  = (const float*)d_in[2];
    const float* Wproj = (const float*)d_in[3];
    const float* bproj = (const float*)d_in[4];
    float* out = (float*)d_out;

    const int M = 2 * 2048;   // B*T = 4096
    const int C = 1024;

    _Float16* xh     = (_Float16*)d_ws;                    // [4096,1024]
    _Float16* Wqkvt  = xh     + (size_t)M * C;             // [3072,1024]
    _Float16* Wprojt = Wqkvt  + (size_t)3 * C * C;         // [1024,1024]
    _Float16* qkvh   = Wprojt + (size_t)C * C;             // [4096,3072]
    _Float16* yh     = qkvh   + (size_t)M * 3 * C;         // [4096,1024]

    // 0) Precision/layout prep.
    cvt_f16_kernel<<<(M * C / 4 + 255) / 256, 256, 0, stream>>>(x, xh, M * C / 4);
    transpose_cvt_kernel<<<dim3(3 * C / 64, C / 64), 256, 0, stream>>>(Wqkv, Wqkvt, C, 3 * C);
    transpose_cvt_kernel<<<dim3(C / 64, C / 64), 256, 0, stream>>>(Wproj, Wprojt, C, C);

    // 1) QKV projection (f16 out).
    gemm_bias_wmma<true><<<dim3(3 * C / 128, M / 128), 256, 0, stream>>>(
        xh, Wqkvt, bqkv, qkvh, M, 3 * C, C);

    // 2) Banded attention: B*H*(T/64) = 1024 blocks.
    attn_band_wmma<<<1024, 128, 0, stream>>>(qkvh, yh);

    // 3) Output projection (f32 out).
    gemm_bias_wmma<false><<<dim3(C / 128, M / 128), 256, 0, stream>>>(
        yh, Wprojt, bproj, out, M, C, C);
}
